// TimeAttentionLayer_7937099563473
// MI455X (gfx1250) — matrix operand order repaired and checked by reading, not compiled, not run on hardware
//
#include <hip/hip_runtime.h>
#include <hip/hip_bf16.h>

typedef __attribute__((ext_vector_type(16))) __bf16 v16bf;
typedef __attribute__((ext_vector_type(8)))  float  v8f;

#define BDIM 64
#define TDIM 4096
#define DDIM 256
#define UDIM 256

// ---------------------------------------------------------------------------
// Prep: Wb = bf16(W1 + W2)  (row-major [K=256][N=256]), and zero context slab.
// Blocks 0..255 -> 65536 W elements; blocks 256..319 -> 16384 context zeros.
// ---------------------------------------------------------------------------
__global__ __launch_bounds__(256)
void prep_kernel(const float* __restrict__ W1, const float* __restrict__ W2,
                 __bf16* __restrict__ Wb, float* __restrict__ ctx) {
    int idx = blockIdx.x * 256 + threadIdx.x;
    if (blockIdx.x < 256) {
        Wb[idx] = (__bf16)(W1[idx] + W2[idx]);
    } else {
        ctx[idx - 256 * 256] = 0.0f;
    }
}

// ---------------------------------------------------------------------------
// Fused GEMM + tanh + dot(v): score[r] = sum_u tanh(x[r,:] . W[:,u]) * v[u]
// One wave handles 16 rows. A (16x256) held in registers as 8 bf16 fragments;
// loop 16 N-tiles x 8 K-steps of v_wmma_f32_16x16x32_bf16 (128 WMMA / wave).
// ---------------------------------------------------------------------------
__global__ __launch_bounds__(256)
void score_kernel(const float* __restrict__ X, const __bf16* __restrict__ Wb,
                  const float* __restrict__ v, float* __restrict__ scoreOut) {
    const int lane = threadIdx.x & 31;
    const int wave = threadIdx.x >> 5;
    const int half = lane >> 4;       // 0: K lo pattern, 1: K hi pattern
    const int m16  = lane & 15;       // row within 16-row tile / N within tile
    const long rowBase = ((long)blockIdx.x * 8 + wave) * 16;
    const float* xrow = X + (rowBase + m16) * (long)DDIM;

    // Load x rows once, convert f32 -> bf16 into WMMA A-fragment layout.
    v16bf a[8];
#pragma unroll
    for (int k = 0; k < 8; ++k) {
        const int c0 = k * 32 + half * 8;
        float4 f0 = *(const float4*)(xrow + c0);
        float4 f1 = *(const float4*)(xrow + c0 + 4);
        float4 f2 = *(const float4*)(xrow + c0 + 16);
        float4 f3 = *(const float4*)(xrow + c0 + 20);
        a[k][0]  = (__bf16)f0.x; a[k][1]  = (__bf16)f0.y;
        a[k][2]  = (__bf16)f0.z; a[k][3]  = (__bf16)f0.w;
        a[k][4]  = (__bf16)f1.x; a[k][5]  = (__bf16)f1.y;
        a[k][6]  = (__bf16)f1.z; a[k][7]  = (__bf16)f1.w;
        a[k][8]  = (__bf16)f2.x; a[k][9]  = (__bf16)f2.y;
        a[k][10] = (__bf16)f2.z; a[k][11] = (__bf16)f2.w;
        a[k][12] = (__bf16)f3.x; a[k][13] = (__bf16)f3.y;
        a[k][14] = (__bf16)f3.z; a[k][15] = (__bf16)f3.w;
    }

    float partial[8];
#pragma unroll
    for (int j = 0; j < 8; ++j) partial[j] = 0.0f;

#pragma unroll
    for (int nt = 0; nt < 16; ++nt) {
        v8f acc = {};
        // B fragment: lane L holds W col N = nt*16 + L%16, K rows in table order.
        const __bf16* wp = Wb + (size_t)(half * 8) * UDIM + nt * 16 + m16;
#pragma unroll
        for (int k = 0; k < 8; ++k) {
            v16bf b;
#pragma unroll
            for (int s = 0; s < 16; ++s)
                b[s] = wp[(size_t)(k * 32 + (s < 8 ? s : s + 8)) * UDIM];
            acc = __builtin_amdgcn_wmma_f32_16x16x32_bf16(
                false, a[k], false, b, (short)0, acc, false, false);
        }
        // C/D layout: lane L elem j == (M = j + 8*half, N = nt*16 + m16)
        const float vn = v[nt * 16 + m16];
#pragma unroll
        for (int j = 0; j < 8; ++j)
            partial[j] += tanhf(acc[j]) * vn;
    }

    // Reduce over the 16 N-lanes within each half (xor stays inside half).
#pragma unroll
    for (int off = 1; off < 16; off <<= 1) {
#pragma unroll
        for (int j = 0; j < 8; ++j)
            partial[j] += __shfl_xor(partial[j], off, 32);
    }
    if (m16 == 0) {
#pragma unroll
        for (int j = 0; j < 8; ++j)
            scoreOut[rowBase + half * 8 + j] = partial[j];
    }
}

// ---------------------------------------------------------------------------
// Softmax over T=4096 per batch, in place on the attention slab of d_out.
// ---------------------------------------------------------------------------
__global__ __launch_bounds__(256)
void softmax_kernel(float* __restrict__ attn) {
    const int b = blockIdx.x;
    const int tid = threadIdx.x;
    float* p = attn + (size_t)b * TDIM;
    __shared__ float red[256];

    float vals[16];
    float m = -3.402823466e38f;
#pragma unroll
    for (int i = 0; i < 16; ++i) {
        vals[i] = p[tid + i * 256];
        m = fmaxf(m, vals[i]);
    }
    red[tid] = m; __syncthreads();
    for (int s = 128; s > 0; s >>= 1) {
        if (tid < s) red[tid] = fmaxf(red[tid], red[tid + s]);
        __syncthreads();
    }
    m = red[0]; __syncthreads();

    float sum = 0.0f;
#pragma unroll
    for (int i = 0; i < 16; ++i) {
        vals[i] = __expf(vals[i] - m);
        sum += vals[i];
    }
    red[tid] = sum; __syncthreads();
    for (int s = 128; s > 0; s >>= 1) {
        if (tid < s) red[tid] += red[tid + s];
        __syncthreads();
    }
    const float inv = 1.0f / red[0];
#pragma unroll
    for (int i = 0; i < 16; ++i)
        p[tid + i * 256] = vals[i] * inv;
}

// ---------------------------------------------------------------------------
// context[b,d] = sum_t attn[b,t] * x[b,t,d]  (memory-bound, coalesced in d,
// T split into 8 chunks accumulated with global_atomic_add_f32).
// ---------------------------------------------------------------------------
__global__ __launch_bounds__(256)
void context_kernel(const float* __restrict__ X, const float* __restrict__ attn,
                    float* __restrict__ ctx) {
    const int b = blockIdx.x;
    const int t0 = blockIdx.y * 512;
    const int d = threadIdx.x;
    const float* xp = X + ((size_t)b * TDIM + t0) * DDIM + d;
    const float* ap = attn + (size_t)b * TDIM + t0;
    float acc = 0.0f;
#pragma unroll 4
    for (int t = 0; t < 512; ++t)
        acc = fmaf(ap[t], xp[(size_t)t * DDIM], acc);
    atomicAdd(&ctx[b * DDIM + d], acc);
}

// ---------------------------------------------------------------------------
extern "C" void kernel_launch(void* const* d_in, const int* in_sizes, int n_in,
                              void* d_out, int out_size, void* d_ws, size_t ws_size,
                              hipStream_t stream) {
    const float* X  = (const float*)d_in[0];   // [64,4096,256]
    const float* W1 = (const float*)d_in[1];   // [256,256]
    const float* W2 = (const float*)d_in[2];   // [256,256]
    const float* v  = (const float*)d_in[3];   // [256,1]

    float* out  = (float*)d_out;
    float* ctx  = out;                    // [64,256]   first in return order
    float* attn = out + BDIM * DDIM;      // [64,4096,1] second

    __bf16* Wb = (__bf16*)d_ws;           // 128 KB staging of bf16(W1+W2)

    prep_kernel<<<320, 256, 0, stream>>>(W1, W2, Wb, ctx);
    score_kernel<<<(BDIM * TDIM) / 128, 256, 0, stream>>>(X, Wb, v, attn);
    softmax_kernel<<<BDIM, 256, 0, stream>>>(attn);
    context_kernel<<<dim3(BDIM, TDIM / 512), 256, 0, stream>>>(X, attn, ctx);
}
